// b_equinet_fstar_4269197492360
// MI455X (gfx1250) — compile-verified, matrix operand order checked by reading, not run on hardware
//
#include <hip/hip_runtime.h>
#include <stdint.h>

#define BATCH 128
#define SRCW  192          // intermediate state is (B,192,192) f32, reinterpreted per layer
#define KDIM  16384
#define NCOL  6400

typedef __attribute__((ext_vector_type(16))) __bf16 v16bf;
typedef __attribute__((ext_vector_type(8)))  float  v8f;

// ---------------------------------------------------------------------------
// Permutation index math (closed form of _perm_idx / _switch_idx, L=6)
// PK: 0 = identity, 1 = butterfly(PL), 2 = switch (3+3 bit transpose)
// ---------------------------------------------------------------------------
template<int PK, int PL>
__device__ __forceinline__ int permf(int n) {
  if (PK == 0) return n;
  else if (PK == 2) return ((n & 7) << 3) | (n >> 3);
  else {
    constexpr int sh    = 6 - PL;          // 2*delta = 1<<sh
    constexpr int delta = 1 << (sh - 1);
    int g = n >> sh;
    int r = n & ((1 << sh) - 1);
    return (g << sh) | ((r & 1) ? delta : 0) | (r >> 1);
  }
}

// One _bmm2 term: O[o][k] += sum_i ( sum_j X[i][j]*R[j][k] ) * Lm[i][o]
template<int DIN, int DOUT>
__device__ __forceinline__ void term_accum(const float (&X)[DIN][DIN],
                                           const float* __restrict__ Lm,
                                           const float* __restrict__ Rm,
                                           float (&O)[DOUT][DOUT]) {
  float tmp[DIN][DOUT];
#pragma unroll
  for (int i = 0; i < DIN; ++i)
#pragma unroll
    for (int k = 0; k < DOUT; ++k) {
      float s = 0.f;
#pragma unroll
      for (int j = 0; j < DIN; ++j) s += X[i][j] * Rm[j * DOUT + k];
      tmp[i][k] = s;
    }
#pragma unroll
  for (int o = 0; o < DOUT; ++o)
#pragma unroll
    for (int k = 0; k < DOUT; ++k) {
      float s = O[o][k];
#pragma unroll
      for (int i = 0; i < DIN; ++i) s += tmp[i][k] * Lm[i * DOUT + o];
      O[o][k] = s;
    }
}

// ---------------------------------------------------------------------------
// Generic butterfly layer. One thread = one (b,p,q) block position.
//  VIN   : read from raw x (B,128,128,2) interleaved re/im
//  DIN==6: merged read (perm applied on the (64,3) interpretation, then merge)
//  RESID : out = (permuted) input + layer(input)
//  UOUT  : _u_layer combination, single real output (B,128,128)
// ---------------------------------------------------------------------------
template<int NB, int DIN, int DOUT, int PK, int PL, bool RESID, bool UOUT, bool VIN>
__global__ __launch_bounds__(256)
void bf_layer(const float* __restrict__ sr, const float* __restrict__ si,
              const float* __restrict__ prm,
              float* __restrict__ dr, float* __restrict__ di) {
  const int flat = blockIdx.x * 256 + threadIdx.x;
  const int q = flat % NB;
  const int p = (flat / NB) % NB;
  const int b = flat / (NB * NB);

  float xr[DIN][DIN], xi[DIN][DIN];

  if constexpr (VIN) {
    const float* xb = sr + (size_t)b * 128 * 128 * 2;
#pragma unroll
    for (int i = 0; i < DIN; ++i)
#pragma unroll
      for (int j = 0; j < DIN; ++j) {
        const float* px = xb + (((2 * p + i) * 128) + (2 * q + j)) * 2;
        xr[i][j] = px[0];
        xi[i][j] = px[1];
      }
  } else {
    int rr[DIN], cc[DIN];
#pragma unroll
    for (int t = 0; t < DIN; ++t) {
      if (DIN == 6) {   // merged view: (p,i) = (2P + I/3, I%3) before permute fusion
        rr[t] = permf<PK, PL>(2 * p + t / 3) * 3 + (t % 3);
        cc[t] = permf<PK, PL>(2 * q + t / 3) * 3 + (t % 3);
      } else {
        rr[t] = permf<PK, PL>(p) * 3 + t;
        cc[t] = permf<PK, PL>(q) * 3 + t;
      }
    }
    const float* br = sr + (size_t)b * SRCW * SRCW;
    const float* bi = si + (size_t)b * SRCW * SRCW;
#pragma unroll
    for (int i = 0; i < DIN; ++i)
#pragma unroll
      for (int j = 0; j < DIN; ++j) {
        xr[i][j] = br[rr[i] * SRCW + cc[j]];
        xi[i][j] = bi[rr[i] * SRCW + cc[j]];
      }
  }

  float o0[DOUT][DOUT];
  float o1[DOUT][DOUT];
#pragma unroll
  for (int a = 0; a < DOUT; ++a)
#pragma unroll
    for (int c = 0; c < DOUT; ++c) {
      if constexpr (RESID) { o0[a][c] = xr[a][c]; o1[a][c] = xi[a][c]; }
      else                 { o0[a][c] = 0.f;      o1[a][c] = 0.f;      }
    }

  constexpr int PST = NB * DIN * DOUT;          // stride between the 8 param mats
  const float* Lb = prm + (size_t)p * DIN * DOUT;
  const float* Rb = prm + (size_t)q * DIN * DOUT;

  if constexpr (!UOUT) {
    // yr = (xr,L0,R0)+(xr,L1,R1)+(xi,L2,R3)+(xi,L3,R2)
    // yi = (xi,L4,R4)+(xi,L5,R5)+(xr,L6,R7)+(xr,L7,R6)
    term_accum<DIN, DOUT>(xr, Lb + 0 * PST, Rb + 0 * PST, o0);
    term_accum<DIN, DOUT>(xr, Lb + 1 * PST, Rb + 1 * PST, o0);
    term_accum<DIN, DOUT>(xi, Lb + 2 * PST, Rb + 3 * PST, o0);
    term_accum<DIN, DOUT>(xi, Lb + 3 * PST, Rb + 2 * PST, o0);
    term_accum<DIN, DOUT>(xi, Lb + 4 * PST, Rb + 4 * PST, o1);
    term_accum<DIN, DOUT>(xi, Lb + 5 * PST, Rb + 5 * PST, o1);
    term_accum<DIN, DOUT>(xr, Lb + 6 * PST, Rb + 7 * PST, o1);
    term_accum<DIN, DOUT>(xr, Lb + 7 * PST, Rb + 6 * PST, o1);
  } else {
    // u_layer: (xr,L0,R0)+(xr,L3,R3)+(xi,L4,R5)+(xi,L7,R6)
    term_accum<DIN, DOUT>(xr, Lb + 0 * PST, Rb + 0 * PST, o0);
    term_accum<DIN, DOUT>(xr, Lb + 3 * PST, Rb + 3 * PST, o0);
    term_accum<DIN, DOUT>(xi, Lb + 4 * PST, Rb + 5 * PST, o0);
    term_accum<DIN, DOUT>(xi, Lb + 7 * PST, Rb + 6 * PST, o0);
  }

  constexpr int DW = NB * DOUT;
  float* drb = dr + (size_t)b * DW * DW;
#pragma unroll
  for (int o = 0; o < DOUT; ++o)
#pragma unroll
    for (int k = 0; k < DOUT; ++k)
      drb[(size_t)(p * DOUT + o) * DW + q * DOUT + k] = o0[o][k];
  if constexpr (!UOUT) {
    float* dib = di + (size_t)b * DW * DW;
#pragma unroll
    for (int o = 0; o < DOUT; ++o)
#pragma unroll
      for (int k = 0; k < DOUT; ++k)
        dib[(size_t)(p * DOUT + o) * DW + q * DOUT + k] = o1[o][k];
  }
}

// ---------------------------------------------------------------------------
// Fold r_index gather + f32->bf16 into the GEMM A operand (4 MB, once).
// ---------------------------------------------------------------------------
__global__ __launch_bounds__(256)
void gather_bf16(const float* __restrict__ y, const int* __restrict__ ridx,
                 __bf16* __restrict__ Abf) {
  const int t = blockIdx.x * 256 + threadIdx.x;
  const int n = t % KDIM;
  const int b = t / KDIM;
  Abf[t] = (__bf16)y[(size_t)b * KDIM + ridx[n]];
}

// Pack two f32 -> one dword of bf16x2 via native fptrunc (RNE); on gfx1250
// this should select the packed hardware bf16 convert.
__device__ __forceinline__ uint32_t pk_bf16(float lo, float hi) {
  union { __bf16 h[2]; uint32_t u; } r;
  r.h[0] = (__bf16)lo;
  r.h[1] = (__bf16)hi;
  return r.u;
}

// Load + convert one 16(rows)x32(K) fragment pair for a 32-wide K step.
__device__ __forceinline__ void load_frags(const __bf16* __restrict__ arow,
                                           const float* __restrict__ brow,
                                           int kb, v16bf& av, v16bf& bv) {
  union UA { uint32_t u[8]; v16bf v; } a, b;
  const uint4 a0 = *(const uint4*)(arow + kb);        // K 0..7  of this lane-half
  const uint4 a1 = *(const uint4*)(arow + kb + 16);   // K 16..23 of this lane-half
  a.u[0] = a0.x; a.u[1] = a0.y; a.u[2] = a0.z; a.u[3] = a0.w;
  a.u[4] = a1.x; a.u[5] = a1.y; a.u[6] = a1.z; a.u[7] = a1.w;

  const float4 f0 = *(const float4*)(brow + kb);
  const float4 f1 = *(const float4*)(brow + kb + 4);
  const float4 f2 = *(const float4*)(brow + kb + 16);
  const float4 f3 = *(const float4*)(brow + kb + 20);
  b.u[0] = pk_bf16(f0.x, f0.y); b.u[1] = pk_bf16(f0.z, f0.w);
  b.u[2] = pk_bf16(f1.x, f1.y); b.u[3] = pk_bf16(f1.z, f1.w);
  b.u[4] = pk_bf16(f2.x, f2.y); b.u[5] = pk_bf16(f2.z, f2.w);
  b.u[6] = pk_bf16(f3.x, f3.y); b.u[7] = pk_bf16(f3.z, f3.w);
  av = a.v;
  bv = b.v;
}

// ---------------------------------------------------------------------------
// NT-GEMM: out[m,n] = sum_k A[m,k] * Cm[n,k], M=128, N=6400, K=16384.
// Block = 8 waves; wave w owns m-tile w (full M), block.x owns one 16-col
// n-tile -> the streamed Cm tile is reused 8x through L0. Cm is converted
// f32->bf16 in registers (single f32 HBM pass, no extra traffic).
// K unrolled by 64 with two independent accumulators to break the
// wmma->wmma RAW chain on C and double load-level parallelism.
// Fragment addressing per gfx1250 16-bit layout: lanes 0-15 hold K 0..7 &
// 16..23, lanes 16-31 hold K 8..15 & 24..31 of their row.
// ---------------------------------------------------------------------------
__global__ __launch_bounds__(256)
void gemm_wmma(const __bf16* __restrict__ Abf,
               const float* __restrict__ Cm,
               float* __restrict__ out) {
  const int lane = threadIdx.x;              // 0..31
  const int m0   = threadIdx.y * 16;         // wave -> m-tile
  const int n0   = blockIdx.x * 16;
  const int half = lane >> 4;                // lane-half K split
  const int r    = lane & 15;
  const int koff = half * 8;

  const __bf16* arow = Abf + (size_t)(m0 + r) * KDIM + koff;
  const float*  brow = Cm  + (size_t)(n0 + r) * KDIM + koff;

  v8f acc0 = {0.f, 0.f, 0.f, 0.f, 0.f, 0.f, 0.f, 0.f};
  v8f acc1 = {0.f, 0.f, 0.f, 0.f, 0.f, 0.f, 0.f, 0.f};

  for (int kb = 0; kb < KDIM; kb += 64) {
    __builtin_prefetch((const void*)(brow + kb + 4096), 0, 1);  // global_prefetch_b8
    v16bf a0, b0, a1, b1;
    load_frags(arow, brow, kb,      a0, b0);
    load_frags(arow, brow, kb + 32, a1, b1);
    acc0 = __builtin_amdgcn_wmma_f32_16x16x32_bf16(false, a0, false, b0,
                                                   (short)0, acc0, false, false);
    acc1 = __builtin_amdgcn_wmma_f32_16x16x32_bf16(false, a1, false, b1,
                                                   (short)0, acc1, false, false);
  }
  v8f acc = acc0 + acc1;

  // D layout: VGPR i -> row m0 + half*8 + i, col n0 + (lane&15)
  float* ob = out + (size_t)(m0 + half * 8) * NCOL + n0 + r;
#pragma unroll
  for (int i = 0; i < 8; ++i) ob[(size_t)i * NCOL] = acc[i];
}

// ---------------------------------------------------------------------------
extern "C" void kernel_launch(void* const* d_in, const int* in_sizes, int n_in,
                              void* d_out, int out_size, void* d_ws, size_t ws_size,
                              hipStream_t stream) {
  (void)in_sizes; (void)n_in; (void)out_size; (void)ws_size;
  const float* x    = (const float*)d_in[0];
  const float* Vp   = (const float*)d_in[1];
  const float* Hp   = (const float*)d_in[2];
  const float* Mp   = (const float*)d_in[3];
  const float* Gp   = (const float*)d_in[4];
  const float* Up   = (const float*)d_in[5];
  const float* Cm   = (const float*)d_in[6];
  const int*   ridx = (const int*)d_in[7];
  float* out = (float*)d_out;

  const size_t SZ = (size_t)BATCH * SRCW * SRCW;      // 4,718,592 f32
  float* Ar = (float*)d_ws;
  float* Ai = Ar + SZ;
  float* Br = Ai + SZ;
  float* Bi = Br + SZ;
  float* yreal = Bi + SZ;                             // (B,16384)
  __bf16* Abf = (__bf16*)(yreal + (size_t)BATCH * KDIM);

  const dim3 blk(256);
  const int g64 = BATCH * 64 * 64 / 256;   // 2048 blocks
  const int g32 = BATCH * 32 * 32 / 256;   // 512 blocks

  // V layer: x -> (B,64,3,64,3)
  bf_layer<64,2,3, 0,0, false,false,true ><<<g64, blk, 0, stream>>>(x, nullptr, Vp, Ar, Ai);
  // H stages: fused pre-permute (l=5,4,3) + merged 6x6 complex layer
  bf_layer<32,6,6, 1,5, false,false,false><<<g32, blk, 0, stream>>>(Ar, Ai, Hp + 0 * 9216, Br, Bi);
  bf_layer<32,6,6, 1,4, false,false,false><<<g32, blk, 0, stream>>>(Br, Bi, Hp + 1 * 9216, Ar, Ai);
  bf_layer<32,6,6, 1,3, false,false,false><<<g32, blk, 0, stream>>>(Ar, Ai, Hp + 2 * 9216, Br, Bi);
  // middle switch permute fused into first residual M block
  bf_layer<64,3,3, 2,0, true ,false,false><<<g64, blk, 0, stream>>>(Br, Bi, Mp + 0 * 4608, Ar, Ai);
  bf_layer<64,3,3, 0,0, true ,false,false><<<g64, blk, 0, stream>>>(Ar, Ai, Mp + 1 * 4608, Br, Bi);
  bf_layer<64,3,3, 0,0, true ,false,false><<<g64, blk, 0, stream>>>(Br, Bi, Mp + 2 * 4608, Ar, Ai);
  // G stages: merged layer, post-permute (l=2,1,0) fused into the NEXT read
  bf_layer<32,6,6, 0,0, false,false,false><<<g32, blk, 0, stream>>>(Ar, Ai, Gp + 0 * 9216, Br, Bi);
  bf_layer<32,6,6, 1,2, false,false,false><<<g32, blk, 0, stream>>>(Br, Bi, Gp + 1 * 9216, Ar, Ai);
  bf_layer<32,6,6, 1,1, false,false,false><<<g32, blk, 0, stream>>>(Ar, Ai, Gp + 2 * 9216, Br, Bi);
  // U layer (consumes perm l=0): -> real y (B,128,128) = (B,16384)
  bf_layer<64,3,2, 1,0, false,true ,false><<<g64, blk, 0, stream>>>(Br, Bi, Up, yreal, nullptr);

  // r_index gather + bf16 convert of GEMM A operand
  gather_bf16<<<(BATCH * KDIM) / 256, blk, 0, stream>>>(yreal, ridx, Abf);

  // big WMMA GEMM: out(128 x 6400)
  gemm_wmma<<<NCOL / 16, dim3(32, 8), 0, stream>>>(Abf, Cm, out);
}
